// DrugSideEffectModel_5282809774108
// MI455X (gfx1250) — matrix operand order repaired and checked by reading, not compiled, not run on hardware
//
#include <hip/hip_runtime.h>
#include <hip/hip_bf16.h>
#include <math.h>

// ---- model dims ----
#define BB      64
#define LLEN    128
#define DDIM    512
#define NHEAD   8
#define DHH     64      // DDIM / NHEAD
#define NLAYER  4
#define FFD     2048
#define COC     8
#define DEC_INN (COC*LLEN*LLEN)   // 131072
#define H1D     512
#define H2D     128
#define BLTOK   (BB*LLEN)         // 8192

typedef __bf16 bf16_t;
typedef __attribute__((ext_vector_type(16))) __bf16 v16bf;
typedef __attribute__((ext_vector_type(8)))  __bf16 v8bf;
typedef __attribute__((ext_vector_type(8)))  float  v8f;

__device__ __forceinline__ v8f wmma_bf16(const v16bf a, const v16bf b, const v8f c) {
    return __builtin_amdgcn_wmma_f32_16x16x32_bf16(false, a, false, b, (short)0, c, false, false);
}

// =====================================================================
// BIG bf16 WMMA GEMM:  C[M,N] = A[M,K] * Bt[N,K]^T + bias
//   requirements: M % 128 == 0, N % 128 == 0, K % 32 == 0.
//   block = 256 thr = 8 waves (4 M-waves x 2 N-waves); block tile 128x128.
//   wave tile 32(M) x 64(N): 2 x 4 = 8 independent accumulators.
//   B panel (128 x 32) staged to LDS with async DMA, double buffered:
//   global_load_async_to_lds_b128 + s_wait_asynccnt overlap DMA with WMMA.
//   A fragments loaded straight from global (L2-resident reuse).
// EPI: 0 none, 1 exact GELU, 2 ReLU. blockIdx.z batches via strides.
// =====================================================================
template<int EPI>
__global__ __launch_bounds__(256) void gemm_bf16_wmma_big(
    const bf16_t* __restrict__ A, int lda, long long strideA,
    const bf16_t* __restrict__ Bt, int ldb, long long strideB,
    const float* __restrict__ bias,
    float* __restrict__ Cf, bf16_t* __restrict__ Cb,
    int ldc, long long strideC,
    int K)
{
    constexpr int LDSROW = 40;                 // 32 data + 8 pad bf16 (20-bank stride)
    __shared__ __align__(32) bf16_t sB[2][128 * LDSROW];

    const int lane = threadIdx.x & 31;
    const int wave = threadIdx.x >> 5;
    const int nr = lane & 15, kh = lane >> 4;
    const long long bz = blockIdx.z;
    const bf16_t* Ab = A  + bz * strideA;
    const bf16_t* Bb = Bt + bz * strideB;
    const int bm0 = blockIdx.y * 128;
    const int bn0 = blockIdx.x * 128;
    const int wm0 = bm0 + (wave >> 1) * 32;    // wave's first M row
    const int wn0 = bn0 + (wave & 1) * 64;     // wave's first N col

    // async staging: B panel = 128 rows x 64B = 512 x 16B chunks, 2 per thread
    const int c0   = threadIdx.x * 2;
    const int row0 = c0 >> 2,        kc0 = (c0 & 3) * 8;
    const int row1 = (c0 + 1) >> 2,  kc1 = ((c0 + 1) & 3) * 8;

    auto stage = [&](int buf, int k0) {
        const bf16_t* g0 = Bb + (long long)(bn0 + row0) * ldb + k0 + kc0;
        const bf16_t* g1 = Bb + (long long)(bn0 + row1) * ldb + k0 + kc1;
        const unsigned l0 = (unsigned)(size_t)&sB[buf][row0 * LDSROW + kc0];
        const unsigned l1 = (unsigned)(size_t)&sB[buf][row1 * LDSROW + kc1];
        asm volatile("global_load_async_to_lds_b128 %0, %1, off"
                     :: "v"(l0), "v"((unsigned long long)(size_t)g0) : "memory");
        asm volatile("global_load_async_to_lds_b128 %0, %1, off"
                     :: "v"(l1), "v"((unsigned long long)(size_t)g1) : "memory");
    };

    v8f acc[2][4] = {};
    const bf16_t* arow0 = Ab + (long long)(wm0 + nr) * lda;
    const bf16_t* arow1 = Ab + (long long)(wm0 + 16 + nr) * lda;

    const int nsteps = K >> 5;
    stage(0, 0);
    for (int i = 0; i < nsteps; ++i) {
        const int k0 = i << 5;
        if (i + 1 < nsteps) {
            stage((i + 1) & 1, k0 + 32);               // overlap next DMA
            asm volatile("s_wait_asynccnt 2" ::: "memory");  // current tile landed
        } else {
            asm volatile("s_wait_asynccnt 0" ::: "memory");
        }
        __syncthreads();
        const bf16_t* sb = sB[i & 1];

        union { v16bf v; v8bf h[2]; } a0, a1;
        a0.h[0] = *(const v8bf*)(arow0 + k0 + 16 * kh);
        a0.h[1] = *(const v8bf*)(arow0 + k0 + 16 * kh + 8);
        a1.h[0] = *(const v8bf*)(arow1 + k0 + 16 * kh);
        a1.h[1] = *(const v8bf*)(arow1 + k0 + 16 * kh + 8);

        #pragma unroll
        for (int nt = 0; nt < 4; ++nt) {
            const int rrow = (wave & 1) * 64 + nt * 16 + nr;
            union { v16bf v; v8bf h[2]; } bf;
            bf.h[0] = *(const v8bf*)(sb + rrow * LDSROW + 16 * kh);
            bf.h[1] = *(const v8bf*)(sb + rrow * LDSROW + 16 * kh + 8);
            acc[0][nt] = wmma_bf16(a0.v, bf.v, acc[0][nt]);
            acc[1][nt] = wmma_bf16(a1.v, bf.v, acc[1][nt]);
        }
        __syncthreads();   // protect buffer we just read from next stage
    }

    #pragma unroll
    for (int mt = 0; mt < 2; ++mt) {
        #pragma unroll
        for (int nt = 0; nt < 4; ++nt) {
            const int n = wn0 + nt * 16 + nr;
            const float bv = bias ? bias[n] : 0.0f;
            #pragma unroll
            for (int r = 0; r < 8; ++r) {
                const int m = wm0 + mt * 16 + 8 * kh + r;
                float v = acc[mt][nt][r] + bv;
                if (EPI == 1) v = 0.5f * v * (1.0f + erff(v * 0.70710678118654752f));
                if (EPI == 2) v = fmaxf(v, 0.0f);
                const long long idx = bz * strideC + (long long)m * ldc + n;
                if (Cf) Cf[idx] = v;
                if (Cb) Cb[idx] = (bf16_t)v;
            }
        }
    }
}

// =====================================================================
// Small generic WMMA GEMM (decoder: M=64). One 16x16 tile per wave.
// =====================================================================
template<int EPI>
__global__ __launch_bounds__(256) void gemm_bf16_wmma(
    const bf16_t* __restrict__ A, int lda,
    const bf16_t* __restrict__ Bt, int ldb,
    const float* __restrict__ bias,
    float* __restrict__ Cf, bf16_t* __restrict__ Cb, int ldc,
    int M, int N, int K)
{
    const int lane = threadIdx.x & 31;
    const int wave = threadIdx.x >> 5;
    const int nr = lane & 15, kh = lane >> 4;
    const int m0 = (blockIdx.y * 2 + (wave >> 2)) * 16;
    const int n0 = (blockIdx.x * 4 + (wave & 3)) * 16;
    if (m0 >= M || n0 >= N) return;

    const bf16_t* arow = A  + (long long)(m0 + nr) * lda;
    const bf16_t* brow = Bt + (long long)(n0 + nr) * ldb;
    v8f acc = {};
    for (int k0 = 0; k0 < K; k0 += 32) {
        union { v16bf v; v8bf h[2]; } a;
        a.h[0] = *(const v8bf*)(arow + k0 + 16 * kh);
        a.h[1] = *(const v8bf*)(arow + k0 + 16 * kh + 8);
        const v16bf bfrag = *(const v16bf*)(brow + k0 + 16 * kh);
        acc = wmma_bf16(a.v, bfrag, acc);
    }
    const float bv = bias ? bias[n0 + nr] : 0.0f;
    #pragma unroll
    for (int r = 0; r < 8; ++r) {
        const int m = m0 + 8 * kh + r;
        float v = acc[r] + bv;
        if (EPI == 1) v = 0.5f * v * (1.0f + erff(v * 0.70710678118654752f));
        if (EPI == 2) v = fmaxf(v, 0.0f);
        const long long idx = (long long)m * ldc + (n0 + nr);
        if (Cf) Cf[idx] = v;
        if (Cb) Cb[idx] = (bf16_t)v;
    }
}

// =====================================================================
// Attention: one workgroup per (batch, head). 8 waves; wave w owns rows
// [16w,16w+16). scores = Q K^T * 0.125 + mask, softmax, ctx = P V.
// =====================================================================
__global__ __launch_bounds__(256) void attention_wmma(
    const bf16_t* __restrict__ qb, const bf16_t* __restrict__ kb,
    const bf16_t* __restrict__ vb, const int* __restrict__ mask,
    bf16_t* __restrict__ ctxb)
{
    __shared__ __align__(32) bf16_t sP[LLEN * LLEN];   // softmaxed probs, 32KB
    __shared__ __align__(32) bf16_t sVt[DHH * LLEN];   // V^T, 16KB
    const int bh = blockIdx.x;
    const int bbi = bh / NHEAD, hh = bh % NHEAD;
    const bf16_t* Q  = qb + (long long)bbi * LLEN * DDIM + hh * DHH;
    const bf16_t* Kp = kb + (long long)bbi * LLEN * DDIM + hh * DHH;
    const bf16_t* Vp = vb + (long long)bbi * LLEN * DDIM + hh * DHH;
    const int lane = threadIdx.x & 31;
    const int wave = threadIdx.x >> 5;
    const int nr = lane & 15, kh = lane >> 4;

    for (int i = threadIdx.x; i < LLEN * DHH; i += 256) {
        const int j = i / DHH, d = i % DHH;
        sVt[d * LLEN + j] = Vp[(long long)j * DDIM + d];
    }

    const int m0 = wave * 16;
    const bf16_t* arow = Q + (long long)(m0 + nr) * DDIM;
    union { v16bf v; v8bf h[2]; } a0, a1;
    a0.h[0] = *(const v8bf*)(arow + 16 * kh);
    a0.h[1] = *(const v8bf*)(arow + 16 * kh + 8);
    a1.h[0] = *(const v8bf*)(arow + 32 + 16 * kh);
    a1.h[1] = *(const v8bf*)(arow + 40 + 16 * kh);

    v8f s[8];
    #pragma unroll
    for (int t = 0; t < 8; ++t) {
        const bf16_t* brow = Kp + (long long)(t * 16 + nr) * DDIM;
        const v16bf b0 = *(const v16bf*)(brow + 16 * kh);
        const v16bf b1 = *(const v16bf*)(brow + 32 + 16 * kh);
        v8f acc = {};
        acc = wmma_bf16(a0.v, b0, acc);
        acc = wmma_bf16(a1.v, b1, acc);
        s[t] = acc;
    }

    float mterm[8];
    #pragma unroll
    for (int t = 0; t < 8; ++t)
        mterm[t] = (1.0f - (float)mask[bbi * LLEN + t * 16 + nr]) * -1e9f;

    #pragma unroll
    for (int r = 0; r < 8; ++r) {
        float vals[8];
        float mx = -3.4e38f;
        #pragma unroll
        for (int t = 0; t < 8; ++t) {
            vals[t] = s[t][r] * 0.125f + mterm[t];
            mx = fmaxf(mx, vals[t]);
        }
        #pragma unroll
        for (int off = 1; off < 16; off <<= 1)
            mx = fmaxf(mx, __shfl_xor(mx, off, 32));
        float sum = 0.0f;
        #pragma unroll
        for (int t = 0; t < 8; ++t) { vals[t] = __expf(vals[t] - mx); sum += vals[t]; }
        #pragma unroll
        for (int off = 1; off < 16; off <<= 1)
            sum += __shfl_xor(sum, off, 32);
        const float inv = 1.0f / sum;
        const int m = m0 + 8 * kh + r;
        #pragma unroll
        for (int t = 0; t < 8; ++t)
            sP[m * LLEN + t * 16 + nr] = (bf16_t)(vals[t] * inv);
    }
    __syncthreads();

    const bf16_t* prow = sP + (m0 + nr) * LLEN;
    #pragma unroll
    for (int nt = 0; nt < 4; ++nt) {
        const int n0 = nt * 16;
        v8f acc = {};
        #pragma unroll
        for (int k0 = 0; k0 < LLEN; k0 += 32) {
            union { v16bf v; v8bf h[2]; } a;
            a.h[0] = *(const v8bf*)(prow + k0 + 16 * kh);
            a.h[1] = *(const v8bf*)(prow + k0 + 16 * kh + 8);
            const v16bf bfrag = *(const v16bf*)(sVt + (n0 + nr) * LLEN + k0 + 16 * kh);
            acc = wmma_bf16(a.v, bfrag, acc);
        }
        #pragma unroll
        for (int r = 0; r < 8; ++r) {
            const int m = m0 + 8 * kh + r;
            ctxb[(long long)(bbi * LLEN + m) * DDIM + hh * DHH + n0 + nr] = (bf16_t)acc[r];
        }
    }
}

// =====================================================================
// embed + LayerNorm : one block per token row (D=512, 256 thr x 2 elems)
// =====================================================================
__global__ __launch_bounds__(256) void embed_ln(
    const int* __restrict__ ids, const float* __restrict__ tok,
    const float* __restrict__ pos, const float* __restrict__ g,
    const float* __restrict__ bta, float* __restrict__ outf,
    bf16_t* __restrict__ outb)
{
    __shared__ float red[256];
    const int row = blockIdx.x;
    const int l = row % LLEN;
    const int t = threadIdx.x;
    const long long id = ids[row];
    float v0 = tok[id * DDIM + t]       + pos[l * DDIM + t];
    float v1 = tok[id * DDIM + 256 + t] + pos[l * DDIM + 256 + t];
    red[t] = v0 + v1;
    __syncthreads();
    for (int s = 128; s > 0; s >>= 1) { if (t < s) red[t] += red[t + s]; __syncthreads(); }
    const float mean = red[0] * (1.0f / DDIM);
    __syncthreads();
    const float d0 = v0 - mean, d1 = v1 - mean;
    red[t] = d0 * d0 + d1 * d1;
    __syncthreads();
    for (int s = 128; s > 0; s >>= 1) { if (t < s) red[t] += red[t + s]; __syncthreads(); }
    const float inv = rsqrtf(red[0] * (1.0f / DDIM) + 1e-12f);
    const float y0 = d0 * inv * g[t] + bta[t];
    const float y1 = d1 * inv * g[256 + t] + bta[256 + t];
    const long long o = (long long)row * DDIM;
    outf[o + t] = y0;       outf[o + 256 + t] = y1;
    outb[o + t] = (bf16_t)y0; outb[o + 256 + t] = (bf16_t)y1;
}

// residual add + LayerNorm
__global__ __launch_bounds__(256) void add_ln(
    const float* __restrict__ in1, const float* __restrict__ in2,
    const float* __restrict__ g, const float* __restrict__ bta,
    float* __restrict__ outf, bf16_t* __restrict__ outb)
{
    __shared__ float red[256];
    const long long row = blockIdx.x;
    const int t = threadIdx.x;
    const float* p1 = in1 + row * DDIM;
    const float* p2 = in2 + row * DDIM;
    float v0 = p1[t] + p2[t];
    float v1 = p1[256 + t] + p2[256 + t];
    red[t] = v0 + v1;
    __syncthreads();
    for (int s = 128; s > 0; s >>= 1) { if (t < s) red[t] += red[t + s]; __syncthreads(); }
    const float mean = red[0] * (1.0f / DDIM);
    __syncthreads();
    const float d0 = v0 - mean, d1 = v1 - mean;
    red[t] = d0 * d0 + d1 * d1;
    __syncthreads();
    for (int s = 128; s > 0; s >>= 1) { if (t < s) red[t] += red[t + s]; __syncthreads(); }
    const float inv = rsqrtf(red[0] * (1.0f / DDIM) + 1e-12f);
    const float y0 = d0 * inv * g[t] + bta[t];
    const float y1 = d1 * inv * g[256 + t] + bta[256 + t];
    const long long o = row * DDIM;
    outf[o + t] = y0;       outf[o + 256 + t] = y1;
    outb[o + t] = (bf16_t)y0; outb[o + 256 + t] = (bf16_t)y1;
}

// 3x3 conv (pad 1) over [B,1,L,L] -> bf16 flattened [B, CO*L*L]
__global__ void conv3x3_flat(const float* __restrict__ inter,
    const float* __restrict__ w, const float* __restrict__ cb,
    bf16_t* __restrict__ flatb)
{
    const long long idx = (long long)blockIdx.x * 256 + threadIdx.x;
    const long long total = (long long)BB * COC * LLEN * LLEN;
    if (idx >= total) return;
    const int j  = (int)(idx % LLEN);
    const int i  = (int)((idx / LLEN) % LLEN);
    const int co = (int)((idx / (LLEN * LLEN)) % COC);
    const int b  = (int)(idx / ((long long)COC * LLEN * LLEN));
    float s = cb[co];
    #pragma unroll
    for (int ki = 0; ki < 3; ++ki) {
        const int ii = i - 1 + ki;
        if (ii < 0 || ii >= LLEN) continue;
        #pragma unroll
        for (int kj = 0; kj < 3; ++kj) {
            const int jj = j - 1 + kj;
            if (jj < 0 || jj >= LLEN) continue;
            s += w[(co * 3 + ki) * 3 + kj] * inter[((long long)b * LLEN + ii) * LLEN + jj];
        }
    }
    flatb[idx] = (bf16_t)s;
}

// fp32 W[R,C] -> bf16 Wt[C,R]; blockIdx.z = layer slab
__global__ void convert_transpose(const float* __restrict__ W,
    bf16_t* __restrict__ Wt, int R, int C)
{
    const long long base = (long long)blockIdx.z * R * C;
    const long long idx = (long long)blockIdx.x * 256 + threadIdx.x;
    if (idx >= (long long)R * C) return;
    const long long r = idx / C, c = idx % C;
    Wt[base + c * R + r] = (bf16_t)W[base + idx];
}

// score[b] = dot(h2[b,:], W3) + b3
__global__ __launch_bounds__(128) void final_score(
    const float* __restrict__ h2f, const float* __restrict__ W3,
    const float* __restrict__ b3, float* __restrict__ out)
{
    __shared__ float red[128];
    const int b = blockIdx.x, t = threadIdx.x;
    red[t] = h2f[b * H2D + t] * W3[t];
    __syncthreads();
    for (int s = 64; s > 0; s >>= 1) { if (t < s) red[t] += red[t + s]; __syncthreads(); }
    if (t == 0) out[b] = red[0] + b3[0];
}

// =====================================================================
// Host orchestration
// =====================================================================
struct EncW {
    const float *Wi, *Wk, *Wo, *Wo2, *Wq, *Wv;
    const float *b1, *b2, *bi, *bk, *bo, *bo2, *bq, *bv, *g1, *g2;
};

extern "C" void kernel_launch(void* const* d_in, const int* in_sizes, int n_in,
                              void* d_out, int out_size, void* d_ws, size_t ws_size,
                              hipStream_t stream)
{
    (void)in_sizes; (void)n_in; (void)out_size; (void)ws_size;
    // jax pytree flatten (alphabetical dict keys at every level):
    // 0 drug, 1 drug_mask, 2 conv_b, 3 conv_w, 4..9 dec{W1,W2,W3,b1,b2,b3},
    // 10..13 emb_drug{b,g,pos,tok}, 14..17 emb_side{...},
    // 18..33 enc_drug{Wi,Wk,Wo,Wo2,Wq,Wv,b1,b2,bi,bk,bo,bo2,bq,bv,g1,g2},
    // 34..49 enc_side{...}, 50 se_mask, 51 side_effect
    const int*   drug      = (const int*)d_in[0];
    const int*   drug_mask = (const int*)d_in[1];
    const float* conv_b    = (const float*)d_in[2];
    const float* conv_w    = (const float*)d_in[3];
    const float* decW1     = (const float*)d_in[4];
    const float* decW2     = (const float*)d_in[5];
    const float* decW3     = (const float*)d_in[6];
    const float* decb1     = (const float*)d_in[7];
    const float* decb2     = (const float*)d_in[8];
    const float* decb3     = (const float*)d_in[9];
    const float* emb_b[2]   = {(const float*)d_in[10], (const float*)d_in[14]};
    const float* emb_g[2]   = {(const float*)d_in[11], (const float*)d_in[15]};
    const float* emb_pos[2] = {(const float*)d_in[12], (const float*)d_in[16]};
    const float* emb_tok[2] = {(const float*)d_in[13], (const float*)d_in[17]};
    EncW enc[2];
    for (int e = 0; e < 2; ++e) {
        const int bse = 18 + 16 * e;
        enc[e].Wi  = (const float*)d_in[bse + 0];  enc[e].Wk  = (const float*)d_in[bse + 1];
        enc[e].Wo  = (const float*)d_in[bse + 2];  enc[e].Wo2 = (const float*)d_in[bse + 3];
        enc[e].Wq  = (const float*)d_in[bse + 4];  enc[e].Wv  = (const float*)d_in[bse + 5];
        enc[e].b1  = (const float*)d_in[bse + 6];  enc[e].b2  = (const float*)d_in[bse + 7];
        enc[e].bi  = (const float*)d_in[bse + 8];  enc[e].bk  = (const float*)d_in[bse + 9];
        enc[e].bo  = (const float*)d_in[bse + 10]; enc[e].bo2 = (const float*)d_in[bse + 11];
        enc[e].bq  = (const float*)d_in[bse + 12]; enc[e].bv  = (const float*)d_in[bse + 13];
        enc[e].g1  = (const float*)d_in[bse + 14]; enc[e].g2  = (const float*)d_in[bse + 15];
    }
    const int* se_mask = (const int*)d_in[50];
    const int* side    = (const int*)d_in[51];
    float* out = (float*)d_out;

    // ---- workspace layout (bump allocator, 256B aligned) ----
    char* ws = (char*)d_ws;
    size_t cur = 0;
    auto takeB = [&](size_t elems) { bf16_t* p = (bf16_t*)(ws + cur);
        cur += (elems * sizeof(bf16_t) + 255) & ~(size_t)255; return p; };
    auto takeF = [&](size_t elems) { float* p = (float*)(ws + cur);
        cur += (elems * sizeof(float) + 255) & ~(size_t)255; return p; };

    bf16_t* wqT[2]; bf16_t* wkT[2]; bf16_t* wvT[2]; bf16_t* woT[2];
    bf16_t* wiT[2]; bf16_t* wo2T[2];
    for (int e = 0; e < 2; ++e) {
        wqT[e]  = takeB((size_t)NLAYER * DDIM * DDIM);
        wkT[e]  = takeB((size_t)NLAYER * DDIM * DDIM);
        wvT[e]  = takeB((size_t)NLAYER * DDIM * DDIM);
        woT[e]  = takeB((size_t)NLAYER * DDIM * DDIM);
        wiT[e]  = takeB((size_t)NLAYER * FFD * DDIM);   // per layer [FF, D]
        wo2T[e] = takeB((size_t)NLAYER * DDIM * FFD);   // per layer [D, FF]
    }
    bf16_t* w1T = takeB((size_t)H1D * DEC_INN);
    bf16_t* w2T = takeB((size_t)H2D * H1D);
    float*  xf  = takeF((size_t)BLTOK * DDIM);
    float*  af  = takeF((size_t)BLTOK * DDIM);
    float*  gf  = takeF((size_t)BLTOK * DDIM);
    bf16_t* xb  = takeB((size_t)BLTOK * DDIM);
    bf16_t* ab  = takeB((size_t)BLTOK * DDIM);
    bf16_t* qbuf = takeB((size_t)BLTOK * DDIM);
    bf16_t* kbuf = takeB((size_t)BLTOK * DDIM);
    bf16_t* vbuf = takeB((size_t)BLTOK * DDIM);
    bf16_t* ctxb = takeB((size_t)BLTOK * DDIM);
    bf16_t* fb   = takeB((size_t)BLTOK * FFD);
    bf16_t* encb[2] = { takeB((size_t)BLTOK * DDIM), takeB((size_t)BLTOK * DDIM) };
    float*  interf = takeF((size_t)BB * LLEN * LLEN);
    bf16_t* flatb  = takeB((size_t)BB * DEC_INN);
    bf16_t* h1b    = takeB((size_t)BB * H1D);
    float*  h2f    = takeF((size_t)BB * H2D);

    const dim3 blk(256);
    auto cgrid = [](long long n, int z) {
        return dim3((unsigned)((n + 255) / 256), 1, (unsigned)z); };

    // ---- weight convert + transpose to bf16 [N,K] ----
    for (int e = 0; e < 2; ++e) {
        convert_transpose<<<cgrid((long long)DDIM * DDIM, NLAYER), blk, 0, stream>>>(enc[e].Wq, wqT[e], DDIM, DDIM);
        convert_transpose<<<cgrid((long long)DDIM * DDIM, NLAYER), blk, 0, stream>>>(enc[e].Wk, wkT[e], DDIM, DDIM);
        convert_transpose<<<cgrid((long long)DDIM * DDIM, NLAYER), blk, 0, stream>>>(enc[e].Wv, wvT[e], DDIM, DDIM);
        convert_transpose<<<cgrid((long long)DDIM * DDIM, NLAYER), blk, 0, stream>>>(enc[e].Wo, woT[e], DDIM, DDIM);
        convert_transpose<<<cgrid((long long)DDIM * FFD, NLAYER), blk, 0, stream>>>(enc[e].Wi, wiT[e], DDIM, FFD);
        convert_transpose<<<cgrid((long long)FFD * DDIM, NLAYER), blk, 0, stream>>>(enc[e].Wo2, wo2T[e], FFD, DDIM);
    }
    convert_transpose<<<cgrid((long long)DEC_INN * H1D, 1), blk, 0, stream>>>(decW1, w1T, DEC_INN, H1D);
    convert_transpose<<<cgrid((long long)H1D * H2D, 1), blk, 0, stream>>>(decW2, w2T, H1D, H2D);

    // ---- encoders ----
    float* enc_out[2] = { out + BB, out + BB + (size_t)BLTOK * DDIM };
    const dim3 gBigD(DDIM / 128, BLTOK / 128, 1);   // 8192 x 512
    const dim3 gBigF(FFD / 128, BLTOK / 128, 1);    // 8192 x 2048
    for (int e = 0; e < 2; ++e) {
        const EncW& E = enc[e];
        const int* ids = e ? side : drug;
        const int* msk = e ? se_mask : drug_mask;
        embed_ln<<<BLTOK, blk, 0, stream>>>(ids, emb_tok[e], emb_pos[e], emb_g[e], emb_b[e], xf, xb);
        for (int l = 0; l < NLAYER; ++l) {
            const size_t wdd = (size_t)l * DDIM * DDIM;
            gemm_bf16_wmma_big<0><<<gBigD, blk, 0, stream>>>(xb, DDIM, 0, wqT[e] + wdd, DDIM, 0,
                E.bq + l * DDIM, nullptr, qbuf, DDIM, 0, DDIM);
            gemm_bf16_wmma_big<0><<<gBigD, blk, 0, stream>>>(xb, DDIM, 0, wkT[e] + wdd, DDIM, 0,
                E.bk + l * DDIM, nullptr, kbuf, DDIM, 0, DDIM);
            gemm_bf16_wmma_big<0><<<gBigD, blk, 0, stream>>>(xb, DDIM, 0, wvT[e] + wdd, DDIM, 0,
                E.bv + l * DDIM, nullptr, vbuf, DDIM, 0, DDIM);
            attention_wmma<<<BB * NHEAD, blk, 0, stream>>>(qbuf, kbuf, vbuf, msk, ctxb);
            gemm_bf16_wmma_big<0><<<gBigD, blk, 0, stream>>>(ctxb, DDIM, 0, woT[e] + wdd, DDIM, 0,
                E.bo + l * DDIM, gf, nullptr, DDIM, 0, DDIM);
            add_ln<<<BLTOK, blk, 0, stream>>>(xf, gf, E.g1 + l * DDIM, E.b1 + l * DDIM, af, ab);
            gemm_bf16_wmma_big<1><<<gBigF, blk, 0, stream>>>(ab, DDIM, 0,
                wiT[e] + (size_t)l * FFD * DDIM, DDIM, 0,
                E.bi + l * FFD, nullptr, fb, FFD, 0, DDIM);
            gemm_bf16_wmma_big<0><<<gBigD, blk, 0, stream>>>(fb, FFD, 0,
                wo2T[e] + (size_t)l * DDIM * FFD, FFD, 0,
                E.bo2 + l * DDIM, gf, nullptr, DDIM, 0, FFD);
            float*  of = (l == NLAYER - 1) ? enc_out[e] : xf;
            bf16_t* ob = (l == NLAYER - 1) ? encb[e] : xb;
            add_ln<<<BLTOK, blk, 0, stream>>>(af, gf, E.g2 + l * DDIM, E.b2 + l * DDIM, of, ob);
        }
    }

    // ---- interaction: per-batch [L,D] x [L,D]^T -> [B,L,L] f32 ----
    const dim3 gI(1, 1, BB);   // one 128x128 block tile per batch
    gemm_bf16_wmma_big<0><<<gI, blk, 0, stream>>>(
        encb[0], DDIM, (long long)LLEN * DDIM,
        encb[1], DDIM, (long long)LLEN * DDIM,
        nullptr, interf, nullptr, LLEN, (long long)LLEN * LLEN, DDIM);

    // ---- conv 3x3 + bias -> bf16 flatten ----
    conv3x3_flat<<<cgrid((long long)BB * COC * LLEN * LLEN, 1), blk, 0, stream>>>(
        interf, conv_w, conv_b, flatb);

    // ---- decoder MLP ----
    const dim3 g1(H1D / 64, BB / 32, 1);
    gemm_bf16_wmma<2><<<g1, blk, 0, stream>>>(flatb, DEC_INN, w1T, DEC_INN,
        decb1, nullptr, h1b, H1D, BB, H1D, DEC_INN);
    const dim3 g2(H2D / 64, BB / 32, 1);
    gemm_bf16_wmma<2><<<g2, blk, 0, stream>>>(h1b, H1D, w2T, H1D,
        decb2, h2f, nullptr, H2D, BB, H2D, H1D);
    final_score<<<BB, 128, 0, stream>>>(h2f, decW3, decb3, out);
}
